// Qwen3_5GatedDeltaNet_62869731278898
// MI455X (gfx1250) — matrix operand order repaired and checked by reading, not compiled, not run on hardware
//
#include <hip/hip_runtime.h>
#include <hip/hip_bf16.h>
#include <math.h>

// ---------------------------------------------------------------------------
// Qwen3.5 GatedDeltaNet forward for MI455X (gfx1250).
// bf16 WMMA GEMMs with TDM (tensor_load_to_lds) staging of the A tile and a
// transposed-in-LDS B tile so all WMMA fragments load as ds_load_b128.
// Register-resident gated delta-rule scan for the recurrence.
// ---------------------------------------------------------------------------

typedef __attribute__((ext_vector_type(16))) __bf16 v16bf;
typedef __attribute__((ext_vector_type(8)))  float  v8f;
typedef unsigned int v4u __attribute__((ext_vector_type(4)));
typedef int          v8i __attribute__((ext_vector_type(8)));
typedef int          v4i __attribute__((ext_vector_type(4)));

#define HK 16
#define HV 32
#define HD 128
#define DMODEL 2048
#define SEQ 2048
#define BATCH 2
#define ROWS (BATCH * SEQ)               // 4096 token rows
#define KEYDIM (HK * HD)                 // 2048
#define VALDIM (HV * HD)                 // 4096
#define CONVDIM (2 * KEYDIM + VALDIM)    // 8192

// ---------------- fp32 -> bf16 convert ----------------
__global__ void k_f32_to_bf16(const float* __restrict__ in,
                              __bf16* __restrict__ out, int n) {
  int i = blockIdx.x * blockDim.x + threadIdx.x;
  if (i < n) out[i] = (__bf16)in[i];
}

// ---------------- bf16 WMMA GEMM: C(MxN) = A(MxK) * B(KxN), row-major ------
// 256 threads = 8 waves; block tile 128x128; wave tile 32x64 (2x4 WMMA accs).
// A tile staged by the Tensor Data Mover (with HW padding to the LDS stride),
// B tile transposed on LDS store so fragments are contiguous b128 loads.
#define BM 128
#define BN 128
#define BK 32
#define APAD 8   // 4 dwords of TDM pad -> row stride BK+APAD = 40 elements

__device__ __forceinline__ void tdm_load_tile_a(const __bf16* gsrc,
                                                unsigned lds_addr, int K) {
  // D# group 0: count=1 | lds_addr | 57-bit global byte address | type=2
  unsigned long long ga = (unsigned long long)(uintptr_t)gsrc;
  v4u g0;
  g0[0] = 1u;                                    // count = 1, user mode
  g0[1] = lds_addr;                              // LDS byte offset
  g0[2] = (unsigned)(ga & 0xFFFFFFFFu);
  g0[3] = (unsigned)((ga >> 32) & 0x01FFFFFFu) | (2u << 30);  // type = 2

  // D# group 1: data_size=2B; pad_enable, pad_interval=16 dwords (code 3),
  // pad_amount=4 dwords (code 3): each 32-elem row padded to 40 elems.
  unsigned td0 = (unsigned)K;        // tensor dim0 (elements)
  unsigned td1 = 1u << 24;           // plenty of rows; no OOB clamp
  v8i g1;
  g1[0] = (int)((1u << 16) | (1u << 20) | (3u << 22) | (3u << 25));
  g1[1] = (int)((td0 & 0xFFFFu) << 16);                 // tensor_dim0 lo16
  g1[2] = (int)((td0 >> 16) | ((td1 & 0xFFFFu) << 16)); // td0 hi16 | td1 lo16
  g1[3] = (int)((td1 >> 16) | ((unsigned)BK << 16));    // td1 hi16 | tile_dim0
  g1[4] = (int)(unsigned)BM;                            // tile_dim1 | tile_dim2=0
  g1[5] = (int)(unsigned)K;                             // tensor_dim0_stride lo32
  g1[6] = 0;
  g1[7] = 0;

  v4i gz = {0, 0, 0, 0};
#if __clang_major__ >= 23
  v8i gz8 = {0, 0, 0, 0, 0, 0, 0, 0};
  __builtin_amdgcn_tensor_load_to_lds(g0, g1, gz, gz, gz8, 0);
#else
  __builtin_amdgcn_tensor_load_to_lds(g0, g1, gz, gz, 0);
#endif
  __builtin_amdgcn_s_wait_tensorcnt(0);
  asm volatile("" ::: "memory");   // LDS written behind the compiler's back
}

__global__ __launch_bounds__(256)
void k_gemm_bf16(const __bf16* __restrict__ A, const __bf16* __restrict__ B,
                 float* __restrict__ C, int M, int N, int K) {
  __shared__ __bf16 As[BM][BK + APAD];   // row-major A tile (TDM-filled)
  __shared__ __bf16 Bst[BN][BK + APAD];  // TRANSPOSED B tile: [col][k]

  const int tid   = threadIdx.x;
  const int wave  = tid >> 5;
  const int lane  = tid & 31;
  const int lrow  = lane & 15;
  const int lhalf = lane >> 4;
  const int wm = (wave & 3) * 32;   // wave row offset inside block tile
  const int wn = (wave >> 2) * 64;  // wave col offset inside block tile
  const int bm = blockIdx.y * BM;
  const int bn = blockIdx.x * BN;

  v8f acc[2][4];
#pragma unroll
  for (int i = 0; i < 2; i++)
#pragma unroll
    for (int j = 0; j < 4; j++)
#pragma unroll
      for (int r = 0; r < 8; r++) acc[i][j][r] = 0.0f;

  // B-tile loader: thread handles k rows (br, br+1), cols bc..bc+7.
  const int br = (tid >> 4) * 2;    // even k row 0..30
  const int bc = (tid & 15) * 8;    // col 0..120

  const unsigned asAddr = (unsigned)(uintptr_t)(void*)&As[0][0];

  for (int k0 = 0; k0 < K; k0 += BK) {
    // ---- stage A tile via Tensor Data Mover (wave 0 issues + waits) ----
    if (wave == 0)
      tdm_load_tile_a(A + (size_t)bm * K + k0, asAddr, K);

    // ---- stage B tile transposed: pack (k,k+1) bf16 pairs into dwords ----
    const __bf16* bg = B + (size_t)(k0 + br) * N + bn + bc;
    if (k0 + BK < K)  // L2 prefetch of next K tile (global_prefetch_b8)
      __builtin_prefetch(bg + (size_t)BK * N, 0, 1);
    int4 r0 = *(const int4*)bg;
    int4 r1 = *(const int4*)(bg + N);
    const unsigned short* p0 = (const unsigned short*)&r0;
    const unsigned short* p1 = (const unsigned short*)&r1;
#pragma unroll
    for (int i = 0; i < 8; i++) {
      unsigned u = (unsigned)p0[i] | ((unsigned)p1[i] << 16);
      *(unsigned*)&Bst[bc + i][br] = u;
    }
    __syncthreads();

    // A fragment: 16x32 bf16. lane L<16: row M=L, K in {0..15};
    // lanes 16..31: K in {16..31}. Contiguous -> 2x ds_load_b128.
    v16bf afrag[2];
#pragma unroll
    for (int i = 0; i < 2; i++) {
      const int row = wm + i * 16 + lrow;
#pragma unroll
      for (int e = 0; e < 16; e++) {
        const int kk = lhalf * 16 + e;
        afrag[i][e] = As[row][kk];
      }
    }
    // B fragment: lane holds column N=lane%16; K = lhalf*16 + e.
    // Transposed tile makes this contiguous -> 2x ds_load_b128.
    v16bf bfrag[4];
#pragma unroll
    for (int j = 0; j < 4; j++) {
      const int col = wn + j * 16 + lrow;
#pragma unroll
      for (int e = 0; e < 16; e++) {
        const int kk = lhalf * 16 + e;
        bfrag[j][e] = Bst[col][kk];
      }
    }
#pragma unroll
    for (int i = 0; i < 2; i++)
#pragma unroll
      for (int j = 0; j < 4; j++)
        acc[i][j] = __builtin_amdgcn_wmma_f32_16x16x32_bf16(
            false, afrag[i], false, bfrag[j], (short)0, acc[i][j], false, false);
    __syncthreads();
  }

  // C/D layout: VGPR r -> M = lhalf*8 + r, N = lane%16.
#pragma unroll
  for (int i = 0; i < 2; i++)
#pragma unroll
    for (int j = 0; j < 4; j++)
#pragma unroll
      for (int r = 0; r < 8; r++) {
        const int m = bm + wm + i * 16 + lhalf * 8 + r;
        const int n = bn + wn + j * 16 + lrow;
        C[(size_t)m * N + n] = acc[i][j][r];
      }
}

// ---------------- tiny b/a projections (N=32 each) ----------------
__global__ void k_ba_gemm(const float* __restrict__ X, const float* __restrict__ Wb,
                          const float* __restrict__ Wa, float* __restrict__ ab) {
  const int row = blockIdx.x;      // 0..ROWS-1
  const int c = threadIdx.x;       // 0..63 (b: 0..31, a: 32..63)
  const float* w = (c < 32) ? Wb : Wa;
  const int cc = c & 31;
  const float* xr = X + (size_t)row * DMODEL;
  float acc = 0.0f;
  for (int k = 0; k < DMODEL; k++) acc = fmaf(xr[k], w[k * 32 + cc], acc);
  ab[(size_t)row * 64 + c] = acc;
}

// ---------------- causal conv1d (K=4) + SiLU ----------------
__global__ __launch_bounds__(256)
void k_conv_silu(const float* __restrict__ x, const float* __restrict__ w,
                 const float* __restrict__ bias, float* __restrict__ y) {
  const int bs = blockIdx.x;
  const int b = bs / SEQ, s = bs % SEQ;
  for (int c = threadIdx.x; c < CONVDIM; c += blockDim.x) {
    float acc = bias[c];
#pragma unroll
    for (int i = 0; i < 4; i++) {
      int t = s - i;
      if (t >= 0) acc = fmaf(w[c * 4 + i], x[((size_t)b * SEQ + t) * CONVDIM + c], acc);
    }
    float sg = 1.0f / (1.0f + __expf(-acc));
    y[(size_t)bs * CONVDIM + c] = acc * sg;  // silu
  }
}

// ---------------- in-place l2norm of q (with 1/sqrt(Dk)) and k ----------------
__global__ void k_l2norm_qk(float* __restrict__ y) {
  const int idx = blockIdx.x;      // ROWS*32 blocks: 16 q heads then 16 k heads
  const int part = idx & 31;
  const int bs = idx >> 5;
  const int isK = part >> 4;
  const int h = part & 15;
  float* p = y + (size_t)bs * CONVDIM + isK * KEYDIM + h * HD;
  const int d = threadIdx.x;       // 0..127
  float v = p[d];
  float sq = v * v;
#pragma unroll
  for (int o = 16; o > 0; o >>= 1) sq += __shfl_xor(sq, o, 32);
  __shared__ float red[4];
  if ((d & 31) == 0) red[d >> 5] = sq;
  __syncthreads();
  float tot = red[0] + red[1] + red[2] + red[3];
  float sc = rsqrtf(tot + 1e-6f);
  if (!isK) sc *= 0.08838834764831845f;  // 1/sqrt(128)
  p[d] = v * sc;
}

// ---------------- g / beta from a, b projections ----------------
__global__ void k_gbeta(const float* __restrict__ ab, const float* __restrict__ A_log,
                        const float* __restrict__ dt_bias,
                        float* __restrict__ g, float* __restrict__ beta) {
  int i = blockIdx.x * blockDim.x + threadIdx.x;
  if (i >= ROWS * HV) return;
  int row = i >> 5, h = i & 31;
  float bv = ab[(size_t)row * 64 + h];
  float av = ab[(size_t)row * 64 + 32 + h];
  beta[i] = 1.0f / (1.0f + __expf(-bv));
  float x = av + dt_bias[h];
  float sp = (x > 20.0f) ? x : log1pf(__expf(x));
  g[i] = -__expf(A_log[h]) * sp;
}

// ---------------- gated delta-rule scan: state in VGPRs ----------------
// 64 blocks (B*Hv); 256 threads: thread = (kh in {0,1}, v in 0..127),
// owns state[k in kh*64 .. +64)[v] (64 floats in registers).
__global__ __launch_bounds__(256)
void k_scan(const float* __restrict__ mixed, const float* __restrict__ g,
            const float* __restrict__ beta, float* __restrict__ core) {
  const int inst = blockIdx.x;
  const int b = inst >> 5;
  const int h = inst & 31;
  const int hk = h >> 1;           // rep = Hv/Hk = 2
  const int t = threadIdx.x;
  const int v = t & 127;
  const int kh = t >> 7;
  const int kb = kh * 64;

  float st[64];
#pragma unroll
  for (int j = 0; j < 64; j++) st[j] = 0.0f;

  __shared__ float q_sh[128], k_sh[128];
  __shared__ float part[2][128];

  for (int s = 0; s < SEQ; s++) {
    const size_t row = (size_t)b * SEQ + s;
    const size_t rb = row * CONVDIM;
    if (t < 128) q_sh[t] = mixed[rb + hk * HD + t];
    else         k_sh[t - 128] = mixed[rb + KEYDIM + hk * HD + (t - 128)];
    __syncthreads();

    const float gv = g[row * HV + h];
    const float bv = beta[row * HV + h];
    const float eg = __expf(gv);

    float partial = 0.0f;                       // decay + k^T * state
#pragma unroll
    for (int j = 0; j < 64; j++) {
      st[j] *= eg;
      partial = fmaf(k_sh[kb + j], st[j], partial);
    }
    part[kh][v] = partial;
    __syncthreads();

    const float kvp   = part[0][v] + part[1][v];
    const float vt    = mixed[rb + 2 * KEYDIM + h * HD + v];
    const float delta = bv * (vt - kvp);

    float op = 0.0f;                            // rank-1 update + q^T * state
#pragma unroll
    for (int j = 0; j < 64; j++) {
      st[j] = fmaf(k_sh[kb + j], delta, st[j]);
      op = fmaf(q_sh[kb + j], st[j], op);
    }
    __syncthreads();
    part[kh][v] = op;
    __syncthreads();
    if (kh == 0)
      core[(row * HV + h) * HD + v] = part[0][v] + part[1][v];
  }
}

// ---------------- per-head RMSNorm * gamma * sigmoid(z) -> bf16 ----------------
__global__ void k_rms_gate(const float* __restrict__ core, const float* __restrict__ z,
                           const float* __restrict__ gamma, __bf16* __restrict__ gated) {
  const int idx = blockIdx.x;      // ROWS*HV
  const int d = threadIdx.x;       // 0..127
  const float c = core[(size_t)idx * HD + d];
  float sq = c * c;
#pragma unroll
  for (int o = 16; o > 0; o >>= 1) sq += __shfl_xor(sq, o, 32);
  __shared__ float red[4];
  if ((d & 31) == 0) red[d >> 5] = sq;
  __syncthreads();
  float mean = (red[0] + red[1] + red[2] + red[3]) * (1.0f / HD);
  float r = c * rsqrtf(mean + 1e-6f);
  float zz = z[(size_t)idx * HD + d];
  float ov = r * gamma[d] * (1.0f / (1.0f + __expf(-zz)));
  gated[(size_t)idx * HD + d] = (__bf16)ov;
}

// ---------------------------------------------------------------------------
extern "C" void kernel_launch(void* const* d_in, const int* in_sizes, int n_in,
                              void* d_out, int out_size, void* d_ws, size_t ws_size,
                              hipStream_t stream) {
  const float* hidden  = (const float*)d_in[0];
  const float* W_qkv   = (const float*)d_in[1];
  const float* W_z     = (const float*)d_in[2];
  const float* W_b     = (const float*)d_in[3];
  const float* W_a     = (const float*)d_in[4];
  const float* conv_w  = (const float*)d_in[5];
  const float* conv_b  = (const float*)d_in[6];
  const float* dt_bias = (const float*)d_in[7];
  const float* A_log   = (const float*)d_in[8];
  const float* gamma   = (const float*)d_in[9];
  const float* W_out   = (const float*)d_in[10];
  float* out = (float*)d_out;
  (void)in_sizes; (void)n_in; (void)out_size; (void)ws_size;

  char* w = (char*)d_ws;
  size_t off = 0;
  auto take = [&](size_t bytes) -> char* {
    char* p = w + off; off += (bytes + 255) & ~(size_t)255; return p;
  };
  __bf16* Xbf     = (__bf16*)take((size_t)ROWS * DMODEL * 2);
  __bf16* Wqkvbf  = (__bf16*)take((size_t)DMODEL * CONVDIM * 2);
  __bf16* Wzbf    = (__bf16*)take((size_t)DMODEL * VALDIM * 2);
  __bf16* Woutbf  = (__bf16*)take((size_t)VALDIM * DMODEL * 2);
  float*  mixedA  = (float*)take((size_t)ROWS * CONVDIM * 4);
  float*  mixedB  = (float*)take((size_t)ROWS * CONVDIM * 4);
  float*  zbuf    = (float*)take((size_t)ROWS * VALDIM * 4);
  float*  ab      = (float*)take((size_t)ROWS * 64 * 4);
  float*  gbuf    = (float*)take((size_t)ROWS * HV * 4);
  float*  betabuf = (float*)take((size_t)ROWS * HV * 4);
  float*  core    = (float*)take((size_t)ROWS * VALDIM * 4);
  __bf16* gated   = (__bf16*)take((size_t)ROWS * VALDIM * 2);

  auto cvt = [&](const float* src, __bf16* dst, int n) {
    k_f32_to_bf16<<<(n + 255) / 256, 256, 0, stream>>>(src, dst, n);
  };
  cvt(hidden, Xbf,    ROWS * DMODEL);
  cvt(W_qkv,  Wqkvbf, DMODEL * CONVDIM);
  cvt(W_z,    Wzbf,   DMODEL * VALDIM);
  cvt(W_out,  Woutbf, VALDIM * DMODEL);

  // QKV and Z projections (bf16 WMMA, f32 accumulate)
  k_gemm_bf16<<<dim3(CONVDIM / BN, ROWS / BM), 256, 0, stream>>>(
      Xbf, Wqkvbf, mixedA, ROWS, CONVDIM, DMODEL);
  k_gemm_bf16<<<dim3(VALDIM / BN, ROWS / BM), 256, 0, stream>>>(
      Xbf, Wzbf, zbuf, ROWS, VALDIM, DMODEL);
  k_ba_gemm<<<ROWS, 64, 0, stream>>>(hidden, W_b, W_a, ab);

  k_conv_silu<<<ROWS, 256, 0, stream>>>(mixedA, conv_w, conv_b, mixedB);
  k_l2norm_qk<<<ROWS * 32, 128, 0, stream>>>(mixedB);
  k_gbeta<<<(ROWS * HV + 255) / 256, 256, 0, stream>>>(ab, A_log, dt_bias, gbuf, betabuf);

  k_scan<<<BATCH * HV, 256, 0, stream>>>(mixedB, gbuf, betabuf, core);

  k_rms_gate<<<ROWS * HV, 128, 0, stream>>>(core, zbuf, gamma, gated);

  // Output projection straight into d_out (f32)
  k_gemm_bf16<<<dim3(DMODEL / BN, ROWS / BM), 256, 0, stream>>>(
      gated, Woutbf, out, ROWS, DMODEL, VALDIM);
}
